// SelfAttention_4243427688600
// MI455X (gfx1250) — matrix operand order repaired and checked by reading, not compiled, not run on hardware
//
#include <hip/hip_runtime.h>
#include <hip/hip_bf16.h>

typedef _Float16 v16h __attribute__((ext_vector_type(16)));
typedef float    v8f  __attribute__((ext_vector_type(8)));
typedef int      v4i  __attribute__((ext_vector_type(4)));

#define HEADS   16
#define HDIM    64
#define SEQ     2048
#define DMODEL  1024
#define BATCH   2
#define ROWS    (BATCH * SEQ)          // 4096

union FragH { v16h v; uint4 q[2]; };
union Acc8  { v8f v; float f[8]; };

// ---------------------------------------------------------------------------
// CDNA5 async global->LDS copies (ASYNCcnt path), guarded so compile never breaks
#if defined(__has_builtin)
#  if __has_builtin(__builtin_amdgcn_global_load_async_to_lds_b128) && \
      __has_builtin(__builtin_amdgcn_s_wait_asynccnt)
#    define USE_ASYNC_LDS 1
#  endif
#endif
#ifndef USE_ASYNC_LDS
#  define USE_ASYNC_LDS 0
#endif

#if USE_ASYNC_LDS
#define AS_GLOBAL __attribute__((address_space(1)))
#define AS_LDS    __attribute__((address_space(3)))
__device__ inline void cp_async_b128(void* lds_dst, const void* gsrc) {
    __builtin_amdgcn_global_load_async_to_lds_b128(
        (AS_GLOBAL v4i*)gsrc, (AS_LDS v4i*)lds_dst, 0, 0);
}
__device__ inline void cp_async_wait() { __builtin_amdgcn_s_wait_asynccnt(0); }
#endif

// ---------------------------------------------------------------------------
// Fragment loaders (layouts per CDNA5 ISA 7.12.2)
// A 16x32 f16: lane L holds row M=L&15; lanes<16: K=0..7,16..23; lanes>=16: K=8..15,24..31
__device__ inline v16h load_afrag(const _Float16* rowp, int lane) {
    FragH u;
    int base = ((lane >> 4) & 1) * 8;
    u.q[0] = *(const uint4*)(rowp + base);
    u.q[1] = *(const uint4*)(rowp + base + 16);
    return u.v;
}
// B 32x16 f16 staged n-major in LDS (row = column n, 32 contiguous K halves)
__device__ inline v16h load_bfrag(const _Float16* rowp, int lane) {
    FragH u;
    int base = ((lane >> 4) & 1) * 8;
    u.q[0] = *(const uint4*)(rowp + base);
    u.q[1] = *(const uint4*)(rowp + base + 16);
    return u.v;
}

// ---------------------------------------------------------------------------
// LayerNorm: one row of d=1024 per 256-thread block, writes f16
__global__ __launch_bounds__(256) void ln_kernel(const float* __restrict__ x,
                                                 const float* __restrict__ g,
                                                 const float* __restrict__ b,
                                                 _Float16* __restrict__ xn) {
    const int row = blockIdx.x;
    const float* xr = x + (size_t)row * DMODEL;
    float s = 0.f, s2 = 0.f;
    for (int i = threadIdx.x; i < DMODEL; i += 256) {
        float v = xr[i]; s += v; s2 += v * v;
    }
    #pragma unroll
    for (int off = 16; off > 0; off >>= 1) {
        s  += __shfl_xor(s,  off, 32);
        s2 += __shfl_xor(s2, off, 32);
    }
    __shared__ float red0[8], red1[8];
    if ((threadIdx.x & 31) == 0) { red0[threadIdx.x >> 5] = s; red1[threadIdx.x >> 5] = s2; }
    __syncthreads();
    float ts = 0.f, ts2 = 0.f;
    #pragma unroll
    for (int i = 0; i < 8; ++i) { ts += red0[i]; ts2 += red1[i]; }
    const float mu   = ts  * (1.0f / DMODEL);
    const float var  = ts2 * (1.0f / DMODEL) - mu * mu;
    const float rinv = rsqrtf(var + 1e-5f);
    _Float16* o = xn + (size_t)row * DMODEL;
    for (int i = threadIdx.x; i < DMODEL; i += 256)
        o[i] = (_Float16)((xr[i] - mu) * rinv * g[i] + b[i]);
}

// ---------------------------------------------------------------------------
// Convert + transpose weights once: W[K][N] f32 -> Wt[N][K] f16 (tiled, LDS)
__global__ __launch_bounds__(256) void cvt_transpose(const float* __restrict__ src,
                                                     _Float16* __restrict__ dst) {
    __shared__ float tile[32][33];
    const int tx = threadIdx.x & 31;
    const int ty = threadIdx.x >> 5;           // 8 row-groups
    const int nb = blockIdx.x * 32;            // N tile
    const int kb = blockIdx.y * 32;            // K tile
    #pragma unroll
    for (int i = ty; i < 32; i += 8)
        tile[i][tx] = src[(size_t)(kb + i) * DMODEL + nb + tx];
    __syncthreads();
    #pragma unroll
    for (int i = ty; i < 32; i += 8)
        dst[(size_t)(nb + i) * DMODEL + kb + tx] = (_Float16)tile[tx][i];
}

// ---------------------------------------------------------------------------
// Tiled WMMA GEMM: C[M,N] = A[M,K] * Bt[N,K]^T, f16 in, f32 acc.
// Block tile 128x128, BK=32 (one WMMA k-step), double-buffered LDS.
// LAYOUT 0: f16 row-major; 1: f32 + bias; 2: f16 head-transposed V^T [b,h,dim,seq]
template<int LAYOUT>
__global__ __launch_bounds__(256) void gemm_wmma(const _Float16* __restrict__ A,
                                                 const _Float16* __restrict__ Bt,
                                                 const float* __restrict__ bias,
                                                 _Float16* __restrict__ Ch,
                                                 float* __restrict__ Cf,
                                                 int M, int N, int K) {
    __shared__ _Float16 As[2][128][32];   // 16 KB
    __shared__ _Float16 Bs[2][128][32];   // 16 KB (n-major)

    const int tid  = threadIdx.x;
    const int lane = tid & 31;
    const int wave = tid >> 5;
    const int wm   = wave >> 2;           // 0..1
    const int wn   = wave & 3;            // 0..3
    const int m0   = blockIdx.y * 128;
    const int n0   = blockIdx.x * 128;

    const int r   = tid >> 1;             // 0..127
    const int off = (tid & 1) * 16;
    const _Float16* aSrc = A  + (size_t)(m0 + r) * K + off;
    const _Float16* bSrc = Bt + (size_t)(n0 + r) * K + off;

    Acc8 acc[4][2];
    #pragma unroll
    for (int i = 0; i < 4; ++i)
        #pragma unroll
        for (int j = 0; j < 2; ++j) acc[i][j].v = (v8f)0.0f;

    // stage tile 0
#if USE_ASYNC_LDS
    cp_async_b128(&As[0][r][off],     aSrc);
    cp_async_b128(&As[0][r][off + 8], aSrc + 8);
    cp_async_b128(&Bs[0][r][off],     bSrc);
    cp_async_b128(&Bs[0][r][off + 8], bSrc + 8);
    cp_async_wait();
#else
    *(uint4*)&As[0][r][off]     = *(const uint4*)(aSrc);
    *(uint4*)&As[0][r][off + 8] = *(const uint4*)(aSrc + 8);
    *(uint4*)&Bs[0][r][off]     = *(const uint4*)(bSrc);
    *(uint4*)&Bs[0][r][off + 8] = *(const uint4*)(bSrc + 8);
#endif
    __syncthreads();

    const int nkt = K / 32;
    for (int kt = 0; kt < nkt; ++kt) {
        const int cur = kt & 1;
        if (kt + 2 < nkt) {
            __builtin_prefetch(aSrc + (kt + 2) * 32, 0, 1);
            __builtin_prefetch(bSrc + (kt + 2) * 32, 0, 1);
        }
#if USE_ASYNC_LDS
        if (kt + 1 < nkt) {
            const _Float16* ap = aSrc + (kt + 1) * 32;
            const _Float16* bp = bSrc + (kt + 1) * 32;
            cp_async_b128(&As[cur ^ 1][r][off],     ap);
            cp_async_b128(&As[cur ^ 1][r][off + 8], ap + 8);
            cp_async_b128(&Bs[cur ^ 1][r][off],     bp);
            cp_async_b128(&Bs[cur ^ 1][r][off + 8], bp + 8);
        }
#else
        uint4 ta0, ta1, tb0, tb1;
        if (kt + 1 < nkt) {
            const _Float16* ap = aSrc + (kt + 1) * 32;
            const _Float16* bp = bSrc + (kt + 1) * 32;
            ta0 = *(const uint4*)(ap);  ta1 = *(const uint4*)(ap + 8);
            tb0 = *(const uint4*)(bp);  tb1 = *(const uint4*)(bp + 8);
        }
#endif
        v16h af[4], bf[2];
        #pragma unroll
        for (int mt = 0; mt < 4; ++mt)
            af[mt] = load_afrag(&As[cur][wm * 64 + mt * 16 + (lane & 15)][0], lane);
        #pragma unroll
        for (int nt = 0; nt < 2; ++nt)
            bf[nt] = load_bfrag(&Bs[cur][wn * 32 + nt * 16 + (lane & 15)][0], lane);

        #pragma unroll
        for (int mt = 0; mt < 4; ++mt)
            #pragma unroll
            for (int nt = 0; nt < 2; ++nt)
                acc[mt][nt].v = __builtin_amdgcn_wmma_f32_16x16x32_f16(
                    false, af[mt], false, bf[nt], (short)0, acc[mt][nt].v, false, false);

#if USE_ASYNC_LDS
        if (kt + 1 < nkt) cp_async_wait();
#else
        if (kt + 1 < nkt) {
            *(uint4*)&As[cur ^ 1][r][off]     = ta0;
            *(uint4*)&As[cur ^ 1][r][off + 8] = ta1;
            *(uint4*)&Bs[cur ^ 1][r][off]     = tb0;
            *(uint4*)&Bs[cur ^ 1][r][off + 8] = tb1;
        }
#endif
        __syncthreads();
    }

    // epilogue (C layout: lane half -> rows r / r+8, N = lane&15)
    const int laneHi = (lane >> 4) & 1;
    #pragma unroll
    for (int mt = 0; mt < 4; ++mt) {
        #pragma unroll
        for (int nt = 0; nt < 2; ++nt) {
            #pragma unroll
            for (int rr = 0; rr < 8; ++rr) {
                const int gr = m0 + wm * 64 + mt * 16 + rr + laneHi * 8;
                const int gc = n0 + wn * 32 + nt * 16 + (lane & 15);
                float v = acc[mt][nt].f[rr];
                if (LAYOUT == 1) {
                    Cf[(size_t)gr * N + gc] = v + bias[gc];
                } else if (LAYOUT == 2) {
                    const int bI = gr >> 11, nn = gr & (SEQ - 1);
                    const int hh = gc >> 6,  dd = gc & (HDIM - 1);
                    Ch[(((size_t)bI * HEADS + hh) * HDIM + dd) * SEQ + nn] = (_Float16)v;
                } else {
                    Ch[(size_t)gr * N + gc] = (_Float16)v;
                }
            }
        }
    }
}

// ---------------------------------------------------------------------------
// RoPE in place on Q and K (first 32 dims of each head), f16 storage
__global__ __launch_bounds__(256) void rope_kernel(_Float16* __restrict__ Q,
                                                   _Float16* __restrict__ K) {
    const long idx = (long)blockIdx.x * 256 + threadIdx.x;    // [0, 2*ROWS*256)
    const int  which = idx >= (long)ROWS * 256;
    const long rem   = idx - (long)which * ROWS * 256;
    const long row   = rem >> 8;                              // flat (b,n)
    const int  li    = (int)(rem & 255);
    const int  h     = li >> 4;
    const int  p     = li & 15;                               // pair index 0..15
    const int  pos   = (int)(row & (SEQ - 1));
    const float inv  = __powf(10000.f, -(2.f * p) / 32.f);
    float sn, cs;
    __sincosf(pos * inv, &sn, &cs);
    _Float16* T = which ? K : Q;
    const size_t off = (size_t)row * DMODEL + h * HDIM + 2 * p;
    const float x0 = (float)T[off], x1 = (float)T[off + 1];
    T[off]     = (_Float16)(x0 * cs - x1 * sn);
    T[off + 1] = (_Float16)(x1 * cs + x0 * sn);
}

// ---------------------------------------------------------------------------
// Flash attention: block = (128 q rows) x (one batch*head). 8 waves x 16 rows.
// Key tiles of 64, double-buffered. V pre-transposed to [b,h,dim,seq].
__global__ __launch_bounds__(256) void attn_kernel(const _Float16* __restrict__ Q,
                                                   const _Float16* __restrict__ K,
                                                   const _Float16* __restrict__ Vt,
                                                   _Float16* __restrict__ AO) {
    __shared__ _Float16 Qs[128][HDIM];        // 16 KB
    __shared__ _Float16 Ks[2][64][HDIM];      // 16 KB (key-major)
    __shared__ _Float16 Vs[2][HDIM][64];      // 16 KB (dim-major)
    __shared__ _Float16 Ps[8][16][64];        // 16 KB (per-wave P slices)

    const int tid  = threadIdx.x;
    const int lane = tid & 31;
    const int wave = tid >> 5;
    const int laneHi = (lane >> 4) & 1;
    const int q0 = blockIdx.x * 128;
    const int bb = blockIdx.y >> 4;
    const int h  = blockIdx.y & 15;
    const size_t base  = (size_t)bb * SEQ * DMODEL + (size_t)h * HDIM;
    const _Float16* Vh = Vt + ((size_t)bb * HEADS + h) * HDIM * SEQ;

    // stage Q tile: 128 x 64 halves, 32 per thread
    {
        const int qr  = tid >> 1;
        const int off = (tid & 1) * 32;
        const _Float16* src = Q + base + (size_t)(q0 + qr) * DMODEL + off;
        #pragma unroll
        for (int j = 0; j < 4; ++j)
            *(uint4*)&Qs[qr][off + 8 * j] = *(const uint4*)(src + 8 * j);
    }

    const int kr   = tid >> 2;            // 0..63  (key row / dim row)
    const int koff = (tid & 3) * 16;      // 0,16,32,48
    const _Float16* kSrc = K + base + (size_t)kr * DMODEL + koff;   // + kt*DMODEL
    const _Float16* vSrc = Vh + (size_t)kr * SEQ + koff;            // + kt

    // stage K/V tile 0
#if USE_ASYNC_LDS
    cp_async_b128(&Ks[0][kr][koff],     kSrc);
    cp_async_b128(&Ks[0][kr][koff + 8], kSrc + 8);
    cp_async_b128(&Vs[0][kr][koff],     vSrc);
    cp_async_b128(&Vs[0][kr][koff + 8], vSrc + 8);
    cp_async_wait();
#else
    *(uint4*)&Ks[0][kr][koff]     = *(const uint4*)(kSrc);
    *(uint4*)&Ks[0][kr][koff + 8] = *(const uint4*)(kSrc + 8);
    *(uint4*)&Vs[0][kr][koff]     = *(const uint4*)(vSrc);
    *(uint4*)&Vs[0][kr][koff + 8] = *(const uint4*)(vSrc + 8);
#endif
    __syncthreads();

    // preload Q A-fragments (2 k-steps over head dim 64)
    v16h qf[2];
    const int qrow = wave * 16 + (lane & 15);
    #pragma unroll
    for (int ks = 0; ks < 2; ++ks) qf[ks] = load_afrag(&Qs[qrow][ks * 32], lane);

    Acc8 o[4];
    #pragma unroll
    for (int dt = 0; dt < 4; ++dt) o[dt].v = (v8f)0.0f;
    float mrow[8], lrow[8];
    #pragma unroll
    for (int rr = 0; rr < 8; ++rr) { mrow[rr] = -1e30f; lrow[rr] = 0.f; }

    const float scale = 0.125f;   // 1/sqrt(64)
    const int nkt = SEQ / 64;

    for (int kt = 0; kt < nkt; ++kt) {
        const int cur = kt & 1;
#if USE_ASYNC_LDS
        if (kt + 1 < nkt) {
            const _Float16* kp = kSrc + (size_t)(kt + 1) * 64 * DMODEL;
            const _Float16* vp = vSrc + (kt + 1) * 64;
            cp_async_b128(&Ks[cur ^ 1][kr][koff],     kp);
            cp_async_b128(&Ks[cur ^ 1][kr][koff + 8], kp + 8);
            cp_async_b128(&Vs[cur ^ 1][kr][koff],     vp);
            cp_async_b128(&Vs[cur ^ 1][kr][koff + 8], vp + 8);
        }
#else
        uint4 tk0, tk1, tv0, tv1;
        if (kt + 1 < nkt) {
            const _Float16* kp = kSrc + (size_t)(kt + 1) * 64 * DMODEL;
            const _Float16* vp = vSrc + (kt + 1) * 64;
            tk0 = *(const uint4*)(kp);  tk1 = *(const uint4*)(kp + 8);
            tv0 = *(const uint4*)(vp);  tv1 = *(const uint4*)(vp + 8);
        }
#endif
        // S = Q K^T : 16 q-rows x 64 keys per wave
        Acc8 s[4];
        #pragma unroll
        for (int nt = 0; nt < 4; ++nt) s[nt].v = (v8f)0.0f;
        #pragma unroll
        for (int nt = 0; nt < 4; ++nt) {
            #pragma unroll
            for (int ks = 0; ks < 2; ++ks) {
                v16h kf = load_bfrag(&Ks[cur][nt * 16 + (lane & 15)][ks * 32], lane);
                s[nt].v = __builtin_amdgcn_wmma_f32_16x16x32_f16(
                    false, qf[ks], false, kf, (short)0, s[nt].v, false, false);
            }
        }

        // online softmax per row (rows r / r+8 in lane halves; width-16 shuffles)
        #pragma unroll
        for (int rr = 0; rr < 8; ++rr) {
            float tm = -1e30f;
            #pragma unroll
            for (int nt = 0; nt < 4; ++nt) tm = fmaxf(tm, s[nt].f[rr] * scale);
            #pragma unroll
            for (int off = 8; off > 0; off >>= 1) tm = fmaxf(tm, __shfl_xor(tm, off, 16));
            const float mnew  = fmaxf(mrow[rr], tm);
            const float alpha = __expf(mrow[rr] - mnew);
            float rs = 0.f;
            #pragma unroll
            for (int nt = 0; nt < 4; ++nt) {
                float p = __expf(s[nt].f[rr] * scale - mnew);
                rs += p;
                Ps[wave][rr + laneHi * 8][nt * 16 + (lane & 15)] = (_Float16)p;
            }
            #pragma unroll
            for (int off = 8; off > 0; off >>= 1) rs += __shfl_xor(rs, off, 16);
            lrow[rr] = lrow[rr] * alpha + rs;
            mrow[rr] = mnew;
            #pragma unroll
            for (int dt = 0; dt < 4; ++dt) o[dt].f[rr] *= alpha;
        }
        // same-wave LDS store->load kept in order by hardware; no barrier needed

        // O += P V : A = P (16x64), B = V^T staged dim-major
        #pragma unroll
        for (int ks = 0; ks < 2; ++ks) {
            v16h pf = load_afrag(&Ps[wave][lane & 15][ks * 32], lane);
            #pragma unroll
            for (int dt = 0; dt < 4; ++dt) {
                v16h vf = load_bfrag(&Vs[cur][dt * 16 + (lane & 15)][ks * 32], lane);
                o[dt].v = __builtin_amdgcn_wmma_f32_16x16x32_f16(
                    false, pf, false, vf, (short)0, o[dt].v, false, false);
            }
        }

#if USE_ASYNC_LDS
        if (kt + 1 < nkt) cp_async_wait();
#else
        if (kt + 1 < nkt) {
            *(uint4*)&Ks[cur ^ 1][kr][koff]     = tk0;
            *(uint4*)&Ks[cur ^ 1][kr][koff + 8] = tk1;
            *(uint4*)&Vs[cur ^ 1][kr][koff]     = tv0;
            *(uint4*)&Vs[cur ^ 1][kr][koff + 8] = tv1;
        }
#endif
        __syncthreads();
    }

    // normalize and write f16 output (layout [b, n, h*64+dim])
    #pragma unroll
    for (int dt = 0; dt < 4; ++dt) {
        #pragma unroll
        for (int rr = 0; rr < 8; ++rr) {
            const int gq  = q0 + wave * 16 + rr + laneHi * 8;
            const int col = dt * 16 + (lane & 15);
            const float v = o[dt].f[rr] / lrow[rr];
            AO[(size_t)(bb * SEQ + gq) * DMODEL + h * HDIM + col] = (_Float16)v;
        }
    }
}

// ---------------------------------------------------------------------------
extern "C" void kernel_launch(void* const* d_in, const int* in_sizes, int n_in,
                              void* d_out, int out_size, void* d_ws, size_t ws_size,
                              hipStream_t stream) {
    const float* x    = (const float*)d_in[0];
    const float* ln_g = (const float*)d_in[1];
    const float* ln_b = (const float*)d_in[2];
    const float* Wq   = (const float*)d_in[3];
    const float* Wk   = (const float*)d_in[4];
    const float* Wv   = (const float*)d_in[5];
    const float* Wo   = (const float*)d_in[6];
    const float* bo   = (const float*)d_in[7];
    float* out = (float*)d_out;

    char* ws = (char*)d_ws;
    const size_t actBytes = (size_t)ROWS * DMODEL * sizeof(_Float16);     // 8 MiB
    const size_t wBytes   = (size_t)DMODEL * DMODEL * sizeof(_Float16);   // 2 MiB
    _Float16* xn16  = (_Float16*)ws;             ws += actBytes;
    _Float16* Wq16t = (_Float16*)ws;             ws += wBytes;
    _Float16* Wk16t = (_Float16*)ws;             ws += wBytes;
    _Float16* Wv16t = (_Float16*)ws;             ws += wBytes;
    _Float16* Wo16t = (_Float16*)ws;             ws += wBytes;
    _Float16* Q16   = (_Float16*)ws;             ws += actBytes;
    _Float16* K16   = (_Float16*)ws;             ws += actBytes;
    _Float16* Vt16  = (_Float16*)ws;             ws += actBytes;   // [b,h,dim,seq]
    _Float16* AO16  = (_Float16*)ws;             ws += actBytes;

    ln_kernel<<<ROWS, 256, 0, stream>>>(x, ln_g, ln_b, xn16);

    dim3 tgrid(DMODEL / 32, DMODEL / 32);   // (32, 32)
    cvt_transpose<<<tgrid, 256, 0, stream>>>(Wq, Wq16t);
    cvt_transpose<<<tgrid, 256, 0, stream>>>(Wk, Wk16t);
    cvt_transpose<<<tgrid, 256, 0, stream>>>(Wv, Wv16t);
    cvt_transpose<<<tgrid, 256, 0, stream>>>(Wo, Wo16t);

    dim3 ggrid(DMODEL / 128, ROWS / 128);   // (8, 32)
    gemm_wmma<0><<<ggrid, 256, 0, stream>>>(xn16, Wq16t, nullptr, Q16, nullptr, ROWS, DMODEL, DMODEL);
    gemm_wmma<0><<<ggrid, 256, 0, stream>>>(xn16, Wk16t, nullptr, K16, nullptr, ROWS, DMODEL, DMODEL);
    gemm_wmma<2><<<ggrid, 256, 0, stream>>>(xn16, Wv16t, nullptr, Vt16, nullptr, ROWS, DMODEL, DMODEL);

    rope_kernel<<<(2 * ROWS * 256) / 256, 256, 0, stream>>>(Q16, K16);

    dim3 agrid(SEQ / 128, BATCH * HEADS);   // (16, 32)
    attn_kernel<<<agrid, 256, 0, stream>>>(Q16, K16, Vt16, AO16);

    gemm_wmma<1><<<ggrid, 256, 0, stream>>>(AO16, Wo16t, bo, nullptr, out, ROWS, DMODEL, DMODEL);
}
